// GGMLMoEGate_26216480375345
// MI455X (gfx1250) — compile-verified, matrix operand order checked by reading, not run on hardware
//
#include <hip/hip_runtime.h>
#include <hip/hip_bf16.h>
#include <stdint.h>

#define HIDDEN    4096
#define N_EXPERTS 64
#define TOP_K     8
#define N_TOKENS  16384
#define N_BLOCKS  1024      // N_EXPERTS*HIDDEN/256
#define BLK_BYTES 144

typedef __attribute__((ext_vector_type(16))) _Float16 v16h;
typedef __attribute__((ext_vector_type(8)))  _Float16 v8h;
typedef __attribute__((ext_vector_type(8)))  float    v8f;
typedef __attribute__((ext_vector_type(4)))  float    v4f;

// ---------------------------------------------------------------------------
// Kernel 1: dequantize Q4_K blocks -> W[64][4096] as f16 (row-major), in d_ws
// One wave32 per 256-element block; lane l handles column l of each 32-wide
// sub-block (4 qs bytes -> 8 output values).
// ---------------------------------------------------------------------------
__global__ __launch_bounds__(256)
void dequant_q4k_kernel(const uint8_t* __restrict__ blocks,
                        _Float16* __restrict__ W) {
  const int gid  = blockIdx.x * blockDim.x + threadIdx.x;
  const int b    = gid >> 5;
  const int lane = gid & 31;
  if (b >= N_BLOCKS) return;

  const uint8_t* blk = blocks + (size_t)b * BLK_BYTES;
  const _Float16* hp = (const _Float16*)blk;      // 2-byte aligned (144*b even)
  const float d    = (float)hp[0];
  const float dmin = (float)hp[1];

  uint8_t s[12];
#pragma unroll
  for (int i = 0; i < 12; ++i) s[i] = blk[4 + i];

  float dsc[8], dmn[8];
#pragma unroll
  for (int i = 0; i < 4; ++i) {
    dsc[i]     = d    * (float)(s[i]     & 63);
    dmn[i]     = dmin * (float)(s[4 + i] & 63);
    dsc[4 + i] = d    * (float)((s[8 + i] & 15) | ((s[i]     >> 2) & 48));
    dmn[4 + i] = dmin * (float)((s[8 + i] >> 4) | ((s[4 + i] >> 2) & 48));
  }

  const int e     = b >> 4;            // 16 blocks per expert row
  const int kbase = (b & 15) * 256;
  _Float16* out = W + (size_t)e * HIDDEN + kbase + lane;

#pragma unroll
  for (int i = 0; i < 4; ++i) {
    const uint8_t q = blk[16 + i * 32 + lane];
    const float lo = dsc[2 * i]     * (float)(q & 15) - dmn[2 * i];
    const float hi = dsc[2 * i + 1] * (float)(q >> 4) - dmn[2 * i + 1];
    out[(2 * i) * 32]     = (_Float16)lo;
    out[(2 * i + 1) * 32] = (_Float16)hi;
  }
}

// ---------------------------------------------------------------------------
// Kernel 2: logits GEMM (WMMA f32<=f16 16x16x32) + softmax + top-8.
// 8 waves/WG; each wave: 32 tokens x 64 experts. Grid = 16384/(8*32) = 64.
// ---------------------------------------------------------------------------
__global__ __launch_bounds__(256)
void moe_gate_kernel(const float* __restrict__ x,
                     const _Float16* __restrict__ W,
                     float* __restrict__ out_w,
                     float* __restrict__ out_i) {
  extern __shared__ float smem[];                 // [8 waves][32 tok][65]
  const int lane = threadIdx.x & 31;
  const int wave = threadIdx.x >> 5;
  const int rif  = lane & 15;                     // row-in-fragment
  const int khalf = (lane >> 4) * 8;              // 0 (lanes 0-15) or 8
  const int t0 = (blockIdx.x * 8 + wave) * 32;

  v8f acc[2][4] = {};

  const float* xrow0 = x + (size_t)(t0 + rif) * HIDDEN;
  const float* xrow1 = x + (size_t)(t0 + 16 + rif) * HIDDEN;
  const _Float16* wr[4];
#pragma unroll
  for (int g = 0; g < 4; ++g)
    wr[g] = W + (size_t)(g * 16 + rif) * HIDDEN;

  for (int k0 = 0; k0 < HIDDEN; k0 += 32) {
    const int ka = k0 + khalf;

    // Prefetch x ~4KB ahead (emits global_prefetch_b8)
    if (k0 + 1024 < HIDDEN) {
      __builtin_prefetch(xrow0 + ka + 1024, 0, 1);
      __builtin_prefetch(xrow1 + ka + 1024, 0, 1);
    }

    // ---- A fragments: fp32 -> f16, ISA 16-bit A layout ----
    v16h a[2];
    {
      const float* xr[2] = {xrow0, xrow1};
#pragma unroll
      for (int mt = 0; mt < 2; ++mt) {
        v4f f0 = __builtin_nontemporal_load((const v4f*)(xr[mt] + ka));
        v4f f1 = __builtin_nontemporal_load((const v4f*)(xr[mt] + ka + 4));
        v4f f2 = __builtin_nontemporal_load((const v4f*)(xr[mt] + ka + 16));
        v4f f3 = __builtin_nontemporal_load((const v4f*)(xr[mt] + ka + 20));
#pragma unroll
        for (int i = 0; i < 4; ++i) {
          a[mt][i]      = (_Float16)f0[i];
          a[mt][4 + i]  = (_Float16)f1[i];
          a[mt][8 + i]  = (_Float16)f2[i];
          a[mt][12 + i] = (_Float16)f3[i];
        }
      }
    }

    // ---- B fragments: L2-resident f16 W, two b128 loads per group ----
    v16h bf[4];
#pragma unroll
    for (int g = 0; g < 4; ++g) {
      v8h b0 = *(const v8h*)(wr[g] + ka);
      v8h b1 = *(const v8h*)(wr[g] + ka + 16);
#pragma unroll
      for (int i = 0; i < 8; ++i) { bf[g][i] = b0[i]; bf[g][8 + i] = b1[i]; }
    }

    // ---- 8 WMMAs per K-step ----
#pragma unroll
    for (int mt = 0; mt < 2; ++mt)
#pragma unroll
      for (int g = 0; g < 4; ++g)
        acc[mt][g] = __builtin_amdgcn_wmma_f32_16x16x32_f16(
            false, a[mt], false, bf[g], (short)0, acc[mt][g], false, false);
  }

  // ---- Scatter logits to LDS: token rows, stride 65 (bank-conflict free) ----
  float* my = smem + (size_t)wave * 32 * 65;
  const int mhi = (lane >= 16) ? 8 : 0;
#pragma unroll
  for (int mt = 0; mt < 2; ++mt)
#pragma unroll
    for (int g = 0; g < 4; ++g)
#pragma unroll
      for (int r = 0; r < 8; ++r) {
        const int m = mt * 16 + r + mhi;
        const int e = g * 16 + (lane & 15);
        my[m * 65 + e] = acc[mt][g][r];
      }
  __syncthreads();

  // ---- Per-lane token: softmax denom + top-8 destructive argmax ----
  float* row = my + lane * 65;
  float mx = row[0];
#pragma unroll
  for (int i = 1; i < N_EXPERTS; ++i) mx = fmaxf(mx, row[i]);
  float denom = 0.f;
#pragma unroll
  for (int i = 0; i < N_EXPERTS; ++i) denom += __expf(row[i] - mx);
  const float inv = 1.0f / denom;

  const int t = t0 + lane;
#pragma unroll
  for (int k = 0; k < TOP_K; ++k) {
    float best = -3.4e38f;
    int bi = 0;
#pragma unroll
    for (int i = 0; i < N_EXPERTS; ++i) {
      const float v = row[i];
      if (v > best) { best = v; bi = i; }   // strict '>': lowest index on ties
    }
    out_w[(size_t)t * TOP_K + k] = __expf(best - mx) * inv;
    out_i[(size_t)t * TOP_K + k] = (float)bi;
    row[bi] = -3.4e38f;
  }
}

// ---------------------------------------------------------------------------
extern "C" void kernel_launch(void* const* d_in, const int* in_sizes, int n_in,
                              void* d_out, int out_size, void* d_ws, size_t ws_size,
                              hipStream_t stream) {
  const float*   x      = (const float*)d_in[0];
  const uint8_t* blocks = (const uint8_t*)d_in[1];

  _Float16* W = (_Float16*)d_ws;                 // 64*4096*2 = 512 KB scratch
  float* out_w = (float*)d_out;
  float* out_i = out_w + (size_t)N_TOKENS * TOP_K;

  // 1) dequantize Q4_K -> f16 W (L2-resident afterwards)
  dequant_q4k_kernel<<<(N_BLOCKS * 32) / 256, 256, 0, stream>>>(blocks, W);

  // 2) GEMM + softmax + top-k
  const size_t lds_bytes = 8u * 32u * 65u * sizeof(float);   // 66,560 B
  moe_gate_kernel<<<N_TOKENS / (8 * 32), 256, lds_bytes, stream>>>(
      x, W, out_w, out_i);
}
